// TransNetExplicit_72705206386720
// MI455X (gfx1250) — matrix operand order repaired and checked by reading, not compiled, not run on hardware
//
#include <hip/hip_runtime.h>
#include <cmath>
#include <stdint.h>

typedef __attribute__((ext_vector_type(16))) _Float16 v16h;
typedef __attribute__((ext_vector_type(8)))  _Float16 v8h;
typedef __attribute__((ext_vector_type(4)))  _Float16 v4h;
typedef __attribute__((ext_vector_type(8)))  float    v8f;

#define BATCH   4096
#define IN_DIM  1024
#define UNITS   2048
#define OUT_DIM 1000
#define NPAD    1024
#define DT_C    0.1f
#define TOL2    1e-8f

// ---------------------------------------------------------------- utilities

__global__ void cvt_f32_to_f16_k(const float* __restrict__ in,
                                 _Float16* __restrict__ out, int n4) {
  int i = blockIdx.x * blockDim.x + threadIdx.x;
  if (i < n4) {
    float4 v = ((const float4*)in)[i];
    v4h o;
    o[0] = (_Float16)v.x; o[1] = (_Float16)v.y;
    o[2] = (_Float16)v.z; o[3] = (_Float16)v.w;
    *(v4h*)(out + (size_t)i * 4) = o;
  }
}

__global__ void copy_f32_k(const float* __restrict__ in,
                           float* __restrict__ out, int n4) {
  int i = blockIdx.x * blockDim.x + threadIdx.x;
  if (i < n4) ((float4*)out)[i] = ((const float4*)in)[i];
}

__global__ void set_int_k(int* p, int v) {
  if (threadIdx.x == 0 && blockIdx.x == 0) *p = v;
}

// in: Kin x Nin row-major f32 ; out: Nout x Kin row-major f16 (grid covers Nout)
__global__ void transpose_cvt_k(const float* __restrict__ in,
                                _Float16* __restrict__ out,
                                int Kin, int Nin) {
  __shared__ float tile[32][33];
  int n0 = blockIdx.x * 32, k0 = blockIdx.y * 32;
  int tx = threadIdx.x, ty = threadIdx.y;      // 32 x 8
#pragma unroll
  for (int j = 0; j < 32; j += 8) {
    int n = n0 + tx, k = k0 + ty + j;
    tile[ty + j][tx] = (n < Nin) ? in[(size_t)k * Nin + n] : 0.f;
  }
  __syncthreads();
#pragma unroll
  for (int j = 0; j < 32; j += 8) {
    int k = k0 + tx, n = n0 + ty + j;
    out[(size_t)n * Kin + k] = (_Float16)tile[tx][ty + j];
  }
}

// ---------------------------------------------------------------- WMMA GEMM
// C = A(MxK f16, row-major) * WT(NxK f16, row-major)^T
// Wave tile 32x64 (c[2][4]); workgroup tile 256x64 (8 waves).
// B tile staged in LDS via CDNA5 async load-to-LDS, double-buffered over
// 128-wide K chunks; A rows are wave-private direct-global loads.
template <int MODE>
__global__ __launch_bounds__(256)
void gemm_wmma_k(const _Float16* __restrict__ A, const _Float16* __restrict__ WT,
                 const float* __restrict__ bias,
                 const float* __restrict__ uin, const float* __restrict__ zcur,
                 float* __restrict__ out32, float* __restrict__ out32b,
                 _Float16* __restrict__ out16,
                 float* __restrict__ partials, const int* __restrict__ done,
                 int M, int N, int K, int Nvalid) {
  // 2 stages x 64 rows x 136 halves (68-dword stride -> conflict-free b128)
  __shared__ _Float16 Blds[2][64][136];

  const int tid  = threadIdx.x;
  const int lane = tid & 31;
  const int wave = tid >> 5;
  const int N0 = blockIdx.x * 64;
  const int M0 = blockIdx.y * 256;

  if (MODE == 1) {
    if (*done) {  // converged: z_next = z_cur (A is the f16 image of zcur, K==N)
      for (int i = tid; i < 256 * 64; i += 256) {
        int r = i >> 6, c = i & 63;
        size_t idx = (size_t)(M0 + r) * N + (N0 + c);
        out32[idx] = zcur[idx];
        out16[idx] = A[idx];
      }
      if (tid == 0) partials[blockIdx.y * gridDim.x + blockIdx.x] = 0.f;
      return;
    }
  }

  // --- async B staging: each thread owns 4 x 16B (row sn, halves [sk,sk+32))
  const int sn = tid >> 2;            // 0..63
  const int sk = (tid & 3) * 32;      // 0,32,64,96
  const _Float16* gB = WT + (size_t)(N0 + sn) * K + sk;
  const uint32_t ldsStage = (uint32_t)(64 * 136 * sizeof(_Float16));
  const uint32_t lbase = (uint32_t)(uintptr_t)&Blds[0][sn][sk];

  auto issue = [&](int kc, int buf) {
    uint64_t g = (uint64_t)(uintptr_t)(gB + kc);
    uint32_t d = lbase + (uint32_t)buf * ldsStage;
    asm volatile(
        "global_load_async_to_lds_b128 %0, %1, off\n\t"
        "global_load_async_to_lds_b128 %0, %1, off offset:16\n\t"
        "global_load_async_to_lds_b128 %0, %1, off offset:32\n\t"
        "global_load_async_to_lds_b128 %0, %1, off offset:48"
        :: "v"(d), "v"(g)
        : "memory");
  };

  issue(0, 0);

  v8f c[2][4];
#pragma unroll
  for (int r = 0; r < 2; r++)
#pragma unroll
    for (int t = 0; t < 4; t++)
#pragma unroll
      for (int e = 0; e < 8; e++) c[r][t][e] = 0.f;

  const int kA = (lane >> 4) << 3;   // 0 or 8
  const int nl = lane & 15;
  const int kb = (lane >> 4) << 3;   // 0 or 8
  const _Float16* Ar0 = A + (size_t)(M0 + wave * 32 + nl) * K + kA;
  const _Float16* Ar1 = Ar0 + (size_t)16 * K;

  auto loadA = [&](const _Float16* p) -> v16h {
    v8h lo = *(const v8h*)p;          // K = kA+0..7
    v8h hi = *(const v8h*)(p + 16);   // K = kA+16..23
    v16h a;
#pragma unroll
    for (int i = 0; i < 8; i++) { a[i] = lo[i]; a[i + 8] = hi[i]; }
    return a;
  };
  auto loadB = [&](int buf, int t, int koff) -> v16h {
    const _Float16* bp = &Blds[buf][nl + t * 16][koff + kb];
    v8h lo = *(const v8h*)bp;
    v8h hi = *(const v8h*)(bp + 16);
    v16h b;
#pragma unroll
    for (int i = 0; i < 8; i++) { b[i] = lo[i]; b[i + 8] = hi[i]; }
    return b;
  };

  for (int kc = 0; kc < K; kc += 128) {
    const int buf = (kc >> 7) & 1;
    if (kc + 128 < K) {
      issue(kc + 128, buf ^ 1);
      asm volatile("s_wait_asynccnt 0x4" ::: "memory");  // buf's loads done
    } else {
      asm volatile("s_wait_asynccnt 0x0" ::: "memory");
    }
    __syncthreads();  // buf visible from every wave's async loads

#pragma unroll
    for (int ks = 0; ks < 4; ++ks) {
      const int k0 = kc + ks * 32;
      v16h a0 = loadA(Ar0 + k0);
      v16h a1 = loadA(Ar1 + k0);
      v16h b0 = loadB(buf, 0, ks * 32);
      v16h b1 = loadB(buf, 1, ks * 32);
      v16h b2 = loadB(buf, 2, ks * 32);
      v16h b3 = loadB(buf, 3, ks * 32);
      c[0][0] = __builtin_amdgcn_wmma_f32_16x16x32_f16(false, a0, false, b0, (short)0, c[0][0], false, false);
      c[1][0] = __builtin_amdgcn_wmma_f32_16x16x32_f16(false, a1, false, b0, (short)0, c[1][0], false, false);
      c[0][1] = __builtin_amdgcn_wmma_f32_16x16x32_f16(false, a0, false, b1, (short)0, c[0][1], false, false);
      c[1][1] = __builtin_amdgcn_wmma_f32_16x16x32_f16(false, a1, false, b1, (short)0, c[1][1], false, false);
      c[0][2] = __builtin_amdgcn_wmma_f32_16x16x32_f16(false, a0, false, b2, (short)0, c[0][2], false, false);
      c[1][2] = __builtin_amdgcn_wmma_f32_16x16x32_f16(false, a1, false, b2, (short)0, c[1][2], false, false);
      c[0][3] = __builtin_amdgcn_wmma_f32_16x16x32_f16(false, a0, false, b3, (short)0, c[0][3], false, false);
      c[1][3] = __builtin_amdgcn_wmma_f32_16x16x32_f16(false, a1, false, b3, (short)0, c[1][3], false, false);
    }
    __syncthreads();  // all waves done reading buf before it is overwritten
  }

  // Epilogue. c[r][t][rr] -> m = M0 + wave*32 + r*16 + rr + 8*(lane>=16),
  //                          n = N0 + (lane&15) + t*16
  const int nB = N0 + nl;
  float errloc = 0.f;
#pragma unroll
  for (int t = 0; t < 4; t++) {
    int n = nB + t * 16;
    float bv = (n < Nvalid) ? bias[n] : 0.f;
#pragma unroll
    for (int r = 0; r < 2; r++) {
      const int mBase = M0 + wave * 32 + r * 16 + ((lane >> 4) << 3);
#pragma unroll
      for (int rr = 0; rr < 8; rr++) {
        int m = mBase + rr;
        size_t idx = (size_t)m * N + n;
        float acc = c[r][t][rr] + bv;
        if (MODE == 0) {
          out32[idx]  = acc;
          out32b[idx] = acc;
          out16[idx]  = (_Float16)acc;
        } else if (MODE == 1) {
          float val = uin[idx] + tanhf(acc) * DT_C;
          float d = val - zcur[idx];
          errloc += d * d;
          out32[idx] = val;
          out16[idx] = (_Float16)val;
        } else {
          if (n < Nvalid) out32[idx] = acc;
        }
      }
    }
  }

  if (MODE == 1) {  // deterministic block reduction of squared residual
    __shared__ float red[256];
    red[tid] = errloc;
    __syncthreads();
    for (int s = 128; s > 0; s >>= 1) {
      if (tid < s) red[tid] += red[tid + s];
      __syncthreads();
    }
    if (tid == 0) partials[blockIdx.y * gridDim.x + blockIdx.x] = red[0];
  }
}

__global__ void reduce_err_k(const float* __restrict__ partials, int n,
                             int* done, float tol2) {
  __shared__ float red[256];
  int tid = threadIdx.x;
  float s = 0.f;
  for (int i = tid; i < n; i += 256) s += partials[i];
  red[tid] = s;
  __syncthreads();
  for (int st = 128; st > 0; st >>= 1) {
    if (tid < st) red[tid] += red[tid + st];
    __syncthreads();
  }
  if (tid == 0 && red[0] < tol2) *done = 1;
}

__global__ void softmax_k(const float* __restrict__ logits,
                          float* __restrict__ out, int ldin, int n) {
  const int row = blockIdx.x;
  const float* L = logits + (size_t)row * ldin;
  __shared__ float red[256];
  int tid = threadIdx.x;
  float mx = -3.402823466e38f;
  for (int i = tid; i < n; i += 256) mx = fmaxf(mx, L[i]);
  red[tid] = mx;
  __syncthreads();
  for (int s = 128; s > 0; s >>= 1) {
    if (tid < s) red[tid] = fmaxf(red[tid], red[tid + s]);
    __syncthreads();
  }
  mx = red[0];
  __syncthreads();
  float sum = 0.f;
  for (int i = tid; i < n; i += 256) sum += expf(L[i] - mx);
  red[tid] = sum;
  __syncthreads();
  for (int s = 128; s > 0; s >>= 1) {
    if (tid < s) red[tid] += red[tid + s];
    __syncthreads();
  }
  float inv = 1.0f / red[0];
  for (int i = tid; i < n; i += 256)
    out[(size_t)row * n + i] = expf(L[i] - mx) * inv;
}

// ---------------------------------------------------------------- host side

extern "C" void kernel_launch(void* const* d_in, const int* in_sizes, int n_in,
                              void* d_out, int out_size, void* d_ws, size_t ws_size,
                              hipStream_t stream) {
  (void)in_sizes; (void)n_in; (void)out_size; (void)ws_size;
  const float* x     = (const float*)d_in[0];
  const float* W_in  = (const float*)d_in[1];
  const float* b_in  = (const float*)d_in[2];
  const float* W_out = (const float*)d_in[3];
  const float* b_out = (const float*)d_in[4];
  const float* Wb[4] = {(const float*)d_in[5], (const float*)d_in[7],
                        (const float*)d_in[9], (const float*)d_in[11]};
  const float* bb[4] = {(const float*)d_in[6], (const float*)d_in[8],
                        (const float*)d_in[10], (const float*)d_in[12]};

  char* ws = (char*)d_ws;
  size_t off = 0;
  auto alloc = [&](size_t bytes) -> char* {
    char* p = ws + off;
    off += (bytes + 255) & ~(size_t)255;
    return p;
  };
  _Float16* xh    = (_Float16*)alloc((size_t)BATCH * IN_DIM * 2);
  _Float16* WTin  = (_Float16*)alloc((size_t)UNITS * IN_DIM * 2);
  _Float16* WT[4];
  for (int i = 0; i < 4; i++) WT[i] = (_Float16*)alloc((size_t)UNITS * UNITS * 2);
  _Float16* WTout = (_Float16*)alloc((size_t)NPAD * UNITS * 2);
  float* uin      = (float*)alloc((size_t)BATCH * UNITS * 4);
  float* z32[2];
  z32[0] = (float*)alloc((size_t)BATCH * UNITS * 4);
  z32[1] = (float*)alloc((size_t)BATCH * UNITS * 4);
  _Float16* z16[2];
  z16[0] = (_Float16*)alloc((size_t)BATCH * UNITS * 2);
  z16[1] = (_Float16*)alloc((size_t)BATCH * UNITS * 2);
  float* logits   = (float*)alloc((size_t)BATCH * NPAD * 4);
  float* partials = (float*)alloc(1024 * 4);
  int* done       = (int*)alloc(256);

  dim3 tb(32, 8);
  cvt_f32_to_f16_k<<<(BATCH * IN_DIM / 4 + 255) / 256, 256, 0, stream>>>(
      x, xh, BATCH * IN_DIM / 4);
  transpose_cvt_k<<<dim3(UNITS / 32, IN_DIM / 32), tb, 0, stream>>>(
      W_in, WTin, IN_DIM, UNITS);
  for (int i = 0; i < 4; i++)
    transpose_cvt_k<<<dim3(UNITS / 32, UNITS / 32), tb, 0, stream>>>(
        Wb[i], WT[i], UNITS, UNITS);
  transpose_cvt_k<<<dim3(NPAD / 32, UNITS / 32), tb, 0, stream>>>(
      W_out, WTout, UNITS, OUT_DIM);

  // z0 = x @ W_in + b_in  -> uin, z32[0], z16[0]
  gemm_wmma_k<0><<<dim3(UNITS / 64, BATCH / 256), 256, 0, stream>>>(
      xh, WTin, b_in, nullptr, nullptr, uin, z32[0], z16[0], nullptr, nullptr,
      BATCH, UNITS, IN_DIM, UNITS);

  int cur = 0;
  for (int blk = 0; blk < 4; ++blk) {
    set_int_k<<<1, 1, 0, stream>>>(done, 0);
    for (int s = 0; s < 50; ++s) {
      gemm_wmma_k<1><<<dim3(UNITS / 64, BATCH / 256), 256, 0, stream>>>(
          z16[cur], WT[blk], bb[blk], uin, z32[cur], z32[cur ^ 1], nullptr,
          z16[cur ^ 1], partials, done, BATCH, UNITS, UNITS, UNITS);
      reduce_err_k<<<1, 256, 0, stream>>>(partials, 512, done, TOL2);
      cur ^= 1;
    }
    if (blk < 3)
      copy_f32_k<<<(BATCH * UNITS / 4 + 255) / 256, 256, 0, stream>>>(
          z32[cur], uin, BATCH * UNITS / 4);
  }

  // logits = z @ W_out + b_out (N padded to 1024, only 1000 valid)
  gemm_wmma_k<2><<<dim3(NPAD / 64, BATCH / 256), 256, 0, stream>>>(
      z16[cur], WTout, b_out, nullptr, nullptr, logits, nullptr, nullptr,
      nullptr, nullptr, BATCH, NPAD, UNITS, OUT_DIM);

  softmax_k<<<BATCH, 256, 0, stream>>>(logits, (float*)d_out, NPAD, OUT_DIM);
}
